// VanillaGNN_56255481643668
// MI455X (gfx1250) — compile-verified, matrix operand order checked by reading, not run on hardware
//
#include <hip/hip_runtime.h>
#include <hip/hip_bf16.h>

// -------------------- types --------------------
typedef __attribute__((ext_vector_type(16))) __bf16       v16bf;
typedef __attribute__((ext_vector_type(8)))  float        v8f;
typedef __attribute__((ext_vector_type(4)))  unsigned int v4u;
typedef __attribute__((ext_vector_type(8)))  int          v8i;
typedef __attribute__((ext_vector_type(4)))  int          v4i;

#define NB 100000
#define ND 200000
#define HH 64

#if __has_builtin(__builtin_amdgcn_tensor_load_to_lds)
#define HAVE_TDM 1
#else
#define HAVE_TDM 0
#endif

// A-fragment K base for 16-bit WMMA 16x32 layout (CDNA5 ISA 7.12.2):
// VGPR j in 0..3 covers K 0..15 (split by lane-half), j in 4..7 covers K 16..31.
__device__ __forceinline__ int kbaseA(int j, int half) {
  return ((j & 4) << 2) + (half << 3) + ((j & 3) << 1);
}

__device__ __forceinline__ unsigned short bfbits(float f) {
  __bf16 b = (__bf16)f;
  unsigned short s;
  __builtin_memcpy(&s, &b, 2);
  return s;
}

// Build bf16 A fragment (16 packed halves = 8 VGPRs) from fp32 row-major data.
__device__ __forceinline__ v16bf frag_from(const float* p, int stride, int row,
                                           int kb, int klim, int half, float scale) {
  v16bf a;
#pragma unroll
  for (int j = 0; j < 8; ++j) {
    int k = kb + kbaseA(j, half);
    float f0 = (k     < klim) ? p[row * stride + k]     * scale : 0.0f;
    float f1 = (k + 1 < klim) ? p[row * stride + k + 1] * scale : 0.0f;
    a[2 * j]     = (__bf16)f0;
    a[2 * j + 1] = (__bf16)f1;
  }
  return a;
}

// TDM: flat 1-D copy of `bytes` (multiple of 8) from global to LDS offset.
// D# per CDNA5 ISA ch.8: group0 {count=1, lds_addr, global_addr, type=2},
// group1 {data_size=8B, tensor_dim0=tile_dim0=bytes/8, tile_dim1=1}.
__device__ __forceinline__ void tdm_load(const void* g, unsigned lds_off, unsigned bytes) {
#if HAVE_TDM
  unsigned long long ga = (unsigned long long)g;
  unsigned n8 = bytes >> 3;
  v4u g0;
  g0[0] = 1u;                                   // count=1, user descriptor
  g0[1] = lds_off;                              // lds_addr (bytes)
  g0[2] = (unsigned)ga;                         // global_addr[31:0]
  g0[3] = (unsigned)((ga >> 32) & 0x1FFFFFFu) | (2u << 30);  // addr[56:32], type=2
  v8i g1;
  g1[0] = (int)(3u << 16);                      // data_size=8B; no mask/pad/iterate
  g1[1] = (int)((n8 & 0xFFFFu) << 16);          // tensor_dim0[15:0] @ bits63:48
  g1[2] = (int)(((n8 >> 16) & 0xFFFFu) | (1u << 16)); // td0[31:16] | tensor_dim1=1
  g1[3] = (int)((n8 & 0xFFFFu) << 16);          // tile_dim0 @ bits127:112
  g1[4] = 1;                                    // tile_dim1=1
  g1[5] = (int)n8;                              // tensor_dim0_stride low32
  g1[6] = 0;
  g1[7] = 0;
  v4i z4 = {0, 0, 0, 0};
#if defined(__clang_major__) && (__clang_major__ >= 23)
  v8i z8 = {0, 0, 0, 0, 0, 0, 0, 0};
  __builtin_amdgcn_tensor_load_to_lds(g0, g1, z4, z4, z8, 0);
#else
  __builtin_amdgcn_tensor_load_to_lds(g0, g1, z4, z4, 0);
#endif
#else
  (void)g; (void)lds_off; (void)bytes;
#endif
}

__device__ __forceinline__ void wait_tensor0() {
#if __has_builtin(__builtin_amdgcn_s_wait_tensorcnt)
  __builtin_amdgcn_s_wait_tensorcnt(0);
#else
  asm volatile("s_wait_tensorcnt 0x0" ::: "memory");
#endif
}

// Pre-pack a fp32 row-major weight matrix W[64][Kw] into the per-lane WMMA
// B-fragment image: uint u -> j=u&7, lane=(u>>3)&31, t=(u>>8)&3, ks=u>>10;
// value = pack_bf16( W[t*16+(lane&15)][ks*32+kbaseA(j,lane>>4)] , +1 ).
__global__ void pack_frag_kernel(const float* W, unsigned* out, int Kw, int nks) {
  int u = blockIdx.x * blockDim.x + threadIdx.x;
  if (u >= nks * 1024) return;
  int j = u & 7, lane = (u >> 3) & 31, t = (u >> 8) & 3, ks = u >> 10;
  int n = t * 16 + (lane & 15);
  int k = ks * 32 + kbaseA(j, lane >> 4);
  float f0 = (k     < Kw) ? W[n * Kw + k]     : 0.0f;
  float f1 = (k + 1 < Kw) ? W[n * Kw + k + 1] : 0.0f;
  out[u] = (unsigned)bfbits(f0) | ((unsigned)bfbits(f1) << 16);
}

// Fused SAGE transform:
//   out[n,:] = act( (sum[n,:]/max(cnt[n],1)) @ Wl^T + x[n,:] @ Wr^T
//                   + bias + (addin ? addin[n,:] : 0) )
// Wlf/Wrf are pre-packed bf16 fragment images, staged into LDS by the TDM.
__global__ void sage_gemm_kernel(const float* sum, const float* cnt,
                                 const unsigned* Wlf, const float* bias,
                                 const float* x, const unsigned* Wrf,
                                 const float* addin, float* out,
                                 int nrows, int K2, int do_relu) {
  __shared__ __attribute__((aligned(32))) unsigned smem_u[4096];  // 16 KB
  const int wave = threadIdx.x >> 5;
  const int lane = threadIdx.x & 31;
  const int base = blockIdx.x * 128 + wave * 16;
  const int nks2 = (K2 + 31) >> 5;

  // LDS byte offset of smem_u: low 32 bits of the generic LDS address
  // (aperture rule, ISA 10.2).
  const unsigned lds_base = (unsigned)(uintptr_t)(void*)smem_u;

  // Stage weight fragments into LDS: Wl image (8KB) at +0, Wr at +8192.
#if HAVE_TDM
  if (wave == 0) {
    if (Wlf) tdm_load(Wlf, lds_base, 8192u);
    tdm_load(Wrf, lds_base + 8192u, (unsigned)(nks2 * 4096));
    wait_tensor0();
  }
#else
  if (Wlf)
    for (int i = threadIdx.x; i < 2048; i += 256) smem_u[i] = Wlf[i];
  for (int i = threadIdx.x; i < nks2 * 1024; i += 256) smem_u[2048 + i] = Wrf[i];
#endif
  __syncthreads();
  // The TDM writes LDS behind the compiler's back (it only sees an integer
  // offset). Escape the pointer so alias analysis cannot fold the ds_loads.
  asm volatile("" : : "r"(&smem_u[0]) : "memory");

  if (base >= nrows) return;  // whole-wave exit; EXEC stays full for WMMA
  const int half = lane >> 4;
  const int nn   = lane & 15;
  const int rowA = base + nn;
  const v16bf* lwl = (const v16bf*)smem_u;
  const v16bf* lwr = (const v16bf*)(smem_u + 2048);

  v8f acc[4];
#pragma unroll
  for (int t = 0; t < 4; ++t)
#pragma unroll
    for (int v = 0; v < 8; ++v) acc[t][v] = 0.0f;

  if (sum) {  // mean-message path, K = 64
    const float inv = 1.0f / fmaxf(cnt[rowA], 1.0f);
#pragma unroll
    for (int ks = 0; ks < 2; ++ks) {
      v16bf a = frag_from(sum, HH, rowA, ks * 32, HH, half, inv);
#pragma unroll
      for (int t = 0; t < 4; ++t) {
        v16bf b = lwl[(ks * 4 + t) * 32 + lane];   // ds_load_b128 x2 from LDS
        acc[t] = __builtin_amdgcn_wmma_f32_16x16x32_bf16(
            false, a, false, b, (short)0, acc[t], false, false);
      }
    }
  }
  if (x) {  // self path, K = K2 (16/32/64)
    for (int ks = 0; ks < nks2; ++ks) {
      v16bf a = frag_from(x, K2, rowA, ks * 32, K2, half, 1.0f);
#pragma unroll
      for (int t = 0; t < 4; ++t) {
        v16bf b = lwr[(ks * 4 + t) * 32 + lane];
        acc[t] = __builtin_amdgcn_wmma_f32_16x16x32_bf16(
            false, a, false, b, (short)0, acc[t], false, false);
      }
    }
  }

  // Epilogue: D layout N = lane&15, M = vgpr + 8*(lane>=16).
#pragma unroll
  for (int t = 0; t < 4; ++t) {
    const int col = t * 16 + nn;
    const float bb = bias ? bias[col] : 0.0f;
#pragma unroll
    for (int v = 0; v < 8; ++v) {
      const int row = base + v + half * 8;
      float r = acc[t][v] + bb;
      if (addin) r += addin[row * HH + col];
      if (do_relu) r = fmaxf(r, 0.0f);
      out[row * HH + col] = r;
    }
  }
}

// Edge gather + segment-sum via fp32 global atomics; lanes cover 32
// consecutive features of one edge -> coalesced gathers and atomics.
__global__ void scatter_kernel(const float* x, const int* src, const int* dst,
                               float* sum, float* cnt, int nedges) {
  const int tid = blockIdx.x * blockDim.x + threadIdx.x;
  if (tid >= nedges * HH) return;
  const int e = tid >> 6, f = tid & 63;
  const int s = src[e], d = dst[e];
  atomicAdd(&sum[d * HH + f], x[s * HH + f]);
  if (f == 0) atomicAdd(&cnt[d], 1.0f);
}

__global__ void colsum_kernel(const float* x, float* colsum, int nrows) {
  const int tid = blockIdx.x * blockDim.x + threadIdx.x;
  const int f = tid & 63;
  const int rstride = (gridDim.x * blockDim.x) >> 6;
  float acc = 0.0f;
  for (int r = tid >> 6; r < nrows; r += rstride) acc += x[r * HH + f];
  atomicAdd(&colsum[f], acc);
}

__global__ void head_kernel(const float* colsum, const float* hW,
                            const float* hb, float* out, float invN) {
  if (threadIdx.x == 0 && blockIdx.x == 0) {
    float s = 0.0f;
    for (int f = 0; f < HH; ++f) s += colsum[f] * invN * hW[f];
    out[0] = s + hb[0];
  }
}

// -------------------- host orchestration --------------------
extern "C" void kernel_launch(void* const* d_in, const int* in_sizes, int n_in,
                              void* d_out, int out_size, void* d_ws, size_t ws_size,
                              hipStream_t stream) {
  (void)in_sizes; (void)n_in; (void)out_size; (void)ws_size;

  const float* bus_x   = (const float*)d_in[0];   // [NB,32]
  const float* dev_x   = (const float*)d_in[1];   // [ND,16]
  const int*   ebb     = (const int*)d_in[2];     // [2,1000000]
  const int*   ebb_src = ebb;
  const int*   ebb_dst = ebb + 1000000;
  const int*   ebd_src = (const int*)d_in[3];
  const int*   ebd_dst = (const int*)d_in[4];
  const int*   edb_src = (const int*)d_in[5];
  const int*   edb_dst = (const int*)d_in[6];
  const float* enc_bus_W = (const float*)d_in[7];   // [64,32]
  const float* enc_bus_b = (const float*)d_in[8];
  const float* enc_dev_W = (const float*)d_in[9];   // [64,16]
  const float* enc_dev_b = (const float*)d_in[10];
  // P[layer][conv: bb=0,bd=1,db=2][Wl=0,bl=1,Wr=2]
  const float* P[2][3][3];
  int idx = 11;
  for (int l = 0; l < 2; ++l)
    for (int c = 0; c < 3; ++c)
      for (int m = 0; m < 3; ++m)
        P[l][c][m] = (const float*)d_in[idx++];
  const float* head_W = (const float*)d_in[idx++];  // [1,64]
  const float* head_b = (const float*)d_in[idx++];  // [1]

  float* ws     = (float*)d_ws;
  float* xbA    = ws; ws += (size_t)NB * HH;
  float* xdA    = ws; ws += (size_t)ND * HH;
  float* xbB    = ws; ws += (size_t)NB * HH;
  float* xdB    = ws; ws += (size_t)ND * HH;
  float* msum   = ws; ws += (size_t)ND * HH;   // reused; NB convs use prefix
  float* mcnt   = ws; ws += (size_t)ND;
  float* colsum = ws; ws += 64;

  // bf16 weight-fragment images (64B aligned)
  unsigned* fbase = (unsigned*)(((uintptr_t)ws + 63) & ~(uintptr_t)63);
  unsigned* f_enc_bus = fbase;            // 1024 uints (nks=1)
  unsigned* f_enc_dev = fbase + 1024;     // 1024 uints
  unsigned* fP[2][3][2];                  // Wl/Wr, 2048 uints each (nks=2)
  unsigned* fp = fbase + 2048;
  for (int l = 0; l < 2; ++l)
    for (int c = 0; c < 3; ++c)
      for (int m = 0; m < 2; ++m) { fP[l][c][m] = fp; fp += 2048; }

  auto pack = [&](const float* W, unsigned* out, int Kw, int nks) {
    pack_frag_kernel<<<(nks * 1024 + 255) / 256, 256, 0, stream>>>(W, out, Kw, nks);
  };
  pack(enc_bus_W, f_enc_bus, 32, 1);
  pack(enc_dev_W, f_enc_dev, 16, 1);
  for (int l = 0; l < 2; ++l)
    for (int c = 0; c < 3; ++c) {
      pack(P[l][c][0], fP[l][c][0], 64, 2);   // Wl
      pack(P[l][c][2], fP[l][c][1], 64, 2);   // Wr
    }

  auto gemm = [&](const float* sum, const float* cnt, const unsigned* Wlf,
                  const float* bias, const float* x, const unsigned* Wrf,
                  const float* addin, float* out, int nrows, int K2, int relu) {
    sage_gemm_kernel<<<(nrows + 127) / 128, 256, 0, stream>>>(
        sum, cnt, Wlf, bias, x, Wrf, addin, out, nrows, K2, relu);
  };
  auto scatter = [&](const float* x, const int* src, const int* dst,
                     int ndst, int nedges) {
    hipMemsetAsync(msum, 0, (size_t)ndst * HH * sizeof(float), stream);
    hipMemsetAsync(mcnt, 0, (size_t)ndst * sizeof(float), stream);
    const int total = nedges * HH;
    scatter_kernel<<<(total + 255) / 256, 256, 0, stream>>>(x, src, dst, msum, mcnt, nedges);
  };

  // Encoders
  gemm(nullptr, nullptr, nullptr, enc_bus_b, bus_x, f_enc_bus, nullptr, xbA, NB, 32, 1);
  gemm(nullptr, nullptr, nullptr, enc_dev_b, dev_x, f_enc_dev, nullptr, xdA, ND, 16, 1);

  auto layer = [&](int l, const float* xbi, const float* xdi, float* xbo, float* xdo) {
    // bd: bus -> device, xd_out = relu(mean@Wl^T + bl + xd@Wr^T)
    scatter(xbi, ebd_src, ebd_dst, ND, 600000);
    gemm(msum, mcnt, fP[l][1][0], P[l][1][1], xdi, fP[l][1][1], nullptr, xdo, ND, 64, 1);
    // bb: bus -> bus, store m_bb (no activation yet)
    scatter(xbi, ebb_src, ebb_dst, NB, 1000000);
    gemm(msum, mcnt, fP[l][0][0], P[l][0][1], xbi, fP[l][0][1], nullptr, xbo, NB, 64, 0);
    // db: device -> bus, xb_out = relu(m_bb + m_db)
    scatter(xdi, edb_src, edb_dst, NB, 600000);
    gemm(msum, mcnt, fP[l][2][0], P[l][2][1], xbi, fP[l][2][1], xbo, xbo, NB, 64, 1);
  };
  layer(0, xbA, xdA, xbB, xdB);
  layer(1, xbB, xdB, xbA, xdA);

  // Readout: mean over bus nodes, then 64-dim head.
  hipMemsetAsync(colsum, 0, 64 * sizeof(float), stream);
  colsum_kernel<<<512, 256, 0, stream>>>(xbA, colsum, NB);
  head_kernel<<<1, 64, 0, stream>>>(colsum, head_W, head_b, (float*)d_out, 1.0f / NB);
}
